// GNN4layer_19542101197605
// MI455X (gfx1250) — matrix operand order repaired and checked by reading, not compiled, not run on hardware
//
#include <hip/hip_runtime.h>
#include <hip/hip_bf16.h>

typedef _Float16 h8   __attribute__((ext_vector_type(8)));
typedef _Float16 v16h __attribute__((ext_vector_type(16)));
typedef float    v8f  __attribute__((ext_vector_type(8)));

#define NN 50000
#define NE 1600000
#define NG 128
#define BN_RSQRT 0.9999950000374997f   /* 1/sqrt(1+1e-5) */
#define NEG_INF_MAPPED 0x007FFFFFu     /* mapOrd(-inf) */

#define CONCAT16(lo, hi) __builtin_shufflevector((lo), (hi), 0,1,2,3,4,5,6,7,8,9,10,11,12,13,14,15)

// Order-preserving float -> u32 map (unsigned compare == float compare)
__device__ __forceinline__ unsigned mapOrd(float x) {
  unsigned b = __float_as_uint(x);
  return ((int)b < 0) ? ~b : (b | 0x80000000u);
}
__device__ __forceinline__ float unmapOrd(unsigned u) {
  unsigned b = (u & 0x80000000u) ? (u ^ 0x80000000u) : ~u;
  return __uint_as_float(b);
}

// ---------------------------------------------------------------------------
// Pack fp32 [K][N] weights into per-lane WMMA B-fragment order (fp16):
// out[((nt*KC + kc)*32 + lane)*16 + h], n = nt*16 + (lane&15),
// k = kc*32 + 8*(lane>>4) + (h&7) + 16*(h>>3). Optional per-row scale (BN fold).
// ---------------------------------------------------------------------------
__global__ void pack_w(const float* __restrict__ W, const float* __restrict__ rowScale,
                       _Float16* __restrict__ out, int Kact, int KP, int N, float scaleMul) {
  int idx = blockIdx.x * blockDim.x + threadIdx.x;
  int KC = KP >> 5;
  int total = (N >> 4) * KC * 512;
  if (idx >= total) return;
  int h    = idx & 15;
  int lane = (idx >> 4) & 31;
  int blk  = idx >> 9;
  int kc   = blk % KC;
  int nt   = blk / KC;
  int n = (nt << 4) + (lane & 15);
  int k = (kc << 5) + ((lane >> 4) << 3) + (h & 7) + ((h >> 3) << 4);
  float v = 0.0f;
  if (k < Kact) {
    v = W[k * N + n];
    if (rowScale) v *= rowScale[k] * scaleMul;
  }
  out[idx] = (_Float16)v;
}

// b2'[n] = b2[n] + sum_k be1[k] * W2[k][n]
__global__ void fold_bias(const float* __restrict__ W2, const float* __restrict__ b2,
                          const float* __restrict__ be1, float* __restrict__ out,
                          int K, int N) {
  int n = blockIdx.x * blockDim.x + threadIdx.x;
  if (n >= N) return;
  float s = b2[n];
  for (int k = 0; k < K; ++k) s += be1[k] * W2[k * N + n];
  out[n] = s;
}

__global__ void fill_u32(unsigned* __restrict__ p, unsigned v, int n) {
  int i = blockIdx.x * blockDim.x + threadIdx.x;
  if (i < n) p[i] = v;
}

// mapped u32 agg -> fp16 node features (-inf / non-finite -> 0)
__global__ void fin_feat(const unsigned* __restrict__ aggm, _Float16* __restrict__ feat, int n) {
  int i = blockIdx.x * blockDim.x + threadIdx.x;
  if (i >= n) return;
  float v = unmapOrd(aggm[i]);
  if (!(v > -3.0e38f)) v = 0.0f;   // catches -inf and NaN
  feat[i] = (_Float16)v;
}

// mapped u32 agg -> fp32 graph features
__global__ void fin_graph(const unsigned* __restrict__ aggm, float* __restrict__ g, int n) {
  int i = blockIdx.x * blockDim.x + threadIdx.x;
  if (i >= n) return;
  float v = unmapOrd(aggm[i]);
  if (!(v > -3.0e38f)) v = 0.0f;
  g[i] = v;
}

// ---------------------------------------------------------------------------
// Fused edge MLP:  64 rows/block, 4 waves, 16 rows per wave.
// MODE 0: conv layer 1 (fp32 node features).  MODE 1: conv layers 2-4 (fp16
// features).  MODE 2: pooling (src = node, pos appended not diffed,
// dst = batch, single GEMM).
// GEMM1: A[16 x K1P] @ W1p -> relu(+b1) -> fp16 LDS
// GEMM2: H[16 x D1]  @ W2p -> relu(+b2')*s2+be2 -> atomic-max scatter
// ---------------------------------------------------------------------------
template <int CIN, int D1, int D2, int MODE>
__global__ __launch_bounds__(128) void edge_mlp(
    const float* __restrict__ xf32, const _Float16* __restrict__ xf16,
    const float* __restrict__ pos,
    const int* __restrict__ srcIdx, const int* __restrict__ dstIdx, int nE,
    const _Float16* __restrict__ W1p, const float* __restrict__ b1,
    const _Float16* __restrict__ W2p, const float* __restrict__ b2f,
    const float* __restrict__ g2, const float* __restrict__ be2,
    unsigned* __restrict__ aggm) {
  constexpr int K1  = CIN + 3;
  constexpr int K1P = (K1 + 31) & ~31;
  constexpr int KC1 = K1P / 32;
  constexpr int KC2 = D1 / 32;
  constexpr int NT1 = D1 / 16;
  constexpr int NT2 = D2 / 16;

  __shared__ __align__(16) _Float16 Ash[64 * K1P];
  __shared__ __align__(16) _Float16 Hsh[(MODE == 2) ? 64 : 64 * D1];
  __shared__ int srcSh[64];
  __shared__ int dstSh[64];

  const int tid  = threadIdx.x;
  const int tile = blockIdx.x * 64;

  // ---- indices ----
  if (tid < 64) {
    int e = tile + tid;
    int s = 0, d = -1;
    if (e < nE) {
      if constexpr (MODE == 2) { s = e; d = dstIdx[e]; }
      else                     { s = srcIdx[e]; d = dstIdx[e]; }
    }
    srcSh[tid] = s;
    dstSh[tid] = d;
  }
  __syncthreads();

  // ---- gather concat(features, pos-part) -> fp16 LDS, zero padded ----
  for (int idx = tid; idx < 64 * K1P; idx += 128) {
    int r = idx / K1P;
    int c = idx - r * K1P;
    float v = 0.0f;
    if (tile + r < nE) {
      int s = srcSh[r];
      if (c < CIN) {
        if constexpr (MODE == 0) v = xf32[s * CIN + c];
        else                     v = (float)xf16[s * CIN + c];
      } else if (c < CIN + 3) {
        int cc = c - CIN;
        float pj = pos[s * 3 + cc];
        if constexpr (MODE == 2) v = pj;
        else                     v = pj - pos[dstSh[r] * 3 + cc];
      }
    }
    Ash[idx] = (_Float16)v;
  }
  __syncthreads();

  const int wave  = tid >> 5;
  const int lane  = tid & 31;
  const int mrow  = lane & 15;
  const int khalf = (lane >> 4) << 3;   // A-fragment K sub-base: 0 or 8
  const int rbase = (lane >> 4) << 3;   // C/D row base: 0 or 8

  const _Float16* Arow = &Ash[(wave * 16 + mrow) * K1P + khalf];

  // ---- GEMM1 ----
  for (int nt = 0; nt < NT1; ++nt) {
    v8f acc = {};
#pragma unroll
    for (int kc = 0; kc < KC1; ++kc) {
      h8 lo = *(const h8*)(Arow + kc * 32);
      h8 hi = *(const h8*)(Arow + kc * 32 + 16);
      v16h a = CONCAT16(lo, hi);
      v16h b = *(const v16h*)(W1p + ((size_t)(nt * KC1 + kc) * 32 + lane) * 16);
      acc = __builtin_amdgcn_wmma_f32_16x16x32_f16(false, a, false, b,
                                                   (short)0, acc, false, false);
    }
    const int cn = nt * 16 + mrow;
    if constexpr (MODE == 2) {
      float bb = b1[cn];
      float ss = g2[cn] * BN_RSQRT;
      float be = be2[cn];
#pragma unroll
      for (int v = 0; v < 8; ++v) {
        int r = wave * 16 + rbase + v;
        int d = dstSh[r];
        if (d >= 0) {
          float h = acc[v] + bb;
          h = h > 0.0f ? h : 0.0f;
          atomicMax(&aggm[(size_t)d * D1 + cn], mapOrd(h * ss + be));
        }
      }
    } else {
      float bb = b1[cn];
#pragma unroll
      for (int v = 0; v < 8; ++v) {
        float h = acc[v] + bb;
        h = h > 0.0f ? h : 0.0f;
        Hsh[(wave * 16 + rbase + v) * D1 + cn] = (_Float16)h;
      }
    }
  }

  if constexpr (MODE != 2) {
    __syncthreads();
    const _Float16* Hrow = &Hsh[(wave * 16 + mrow) * D1 + khalf];
    // ---- GEMM2 ----
    for (int nt = 0; nt < NT2; ++nt) {
      v8f acc = {};
#pragma unroll
      for (int kc = 0; kc < KC2; ++kc) {
        h8 lo = *(const h8*)(Hrow + kc * 32);
        h8 hi = *(const h8*)(Hrow + kc * 32 + 16);
        v16h a = CONCAT16(lo, hi);
        v16h b = *(const v16h*)(W2p + ((size_t)(nt * KC2 + kc) * 32 + lane) * 16);
        acc = __builtin_amdgcn_wmma_f32_16x16x32_f16(false, a, false, b,
                                                     (short)0, acc, false, false);
      }
      const int cn = nt * 16 + mrow;
      float bb = b2f[cn];
      float ss = g2[cn] * BN_RSQRT;
      float be = be2[cn];
#pragma unroll
      for (int v = 0; v < 8; ++v) {
        int r = wave * 16 + rbase + v;
        int d = dstSh[r];
        if (d >= 0) {
          float h = acc[v] + bb;
          h = h > 0.0f ? h : 0.0f;
          atomicMax(&aggm[(size_t)d * D2 + cn], mapOrd(h * ss + be));
        }
      }
    }
  }
}

// ---------------------------------------------------------------------------
// FC head: g[128,128] -> relu(g@W1+b1)*s+be -> @W2+b2 -> out[128,3]
// ---------------------------------------------------------------------------
__global__ __launch_bounds__(256) void head_kernel(
    const float* __restrict__ g,
    const float* __restrict__ fW1, const float* __restrict__ fb1,
    const float* __restrict__ fg1, const float* __restrict__ fbe1,
    const float* __restrict__ fW2, const float* __restrict__ fb2,
    float* __restrict__ out) {
  __shared__ float fsh[128 * 64];
  int tid = threadIdx.x;
  for (int idx = tid; idx < 128 * 64; idx += 256) {
    int r = idx >> 6, c = idx & 63;
    float s = fb1[c];
    for (int k = 0; k < 128; ++k) s += g[r * 128 + k] * fW1[k * 64 + c];
    s = s > 0.0f ? s : 0.0f;
    fsh[idx] = s * (fg1[c] * BN_RSQRT) + fbe1[c];
  }
  __syncthreads();
  for (int idx = tid; idx < 128 * 3; idx += 256) {
    int r = idx / 3, c = idx - r * 3;
    float s = fb2[c];
    for (int k = 0; k < 64; ++k) s += fsh[r * 64 + k] * fW2[k * 3 + c];
    out[idx] = s;
  }
}

// ---------------------------------------------------------------------------
extern "C" void kernel_launch(void* const* d_in, const int* in_sizes, int n_in,
                              void* d_out, int out_size, void* d_ws, size_t ws_size,
                              hipStream_t stream) {
  (void)in_sizes; (void)n_in; (void)out_size; (void)ws_size;

  const float* x     = (const float*)d_in[0];
  const float* pos   = (const float*)d_in[1];
  const int*   batch = (const int*)d_in[2];
  const int*   ei    = (const int*)d_in[3];
  const int* src = ei;
  const int* dst = ei + NE;

  // params: conv l (l=0..3) base 8l: W1,b1,g1,be1,W2,b2,g2,be2 ;
  // 32..35: p_W1,p_b1,p_g1,p_be1 ; 36..41: f_W1,f_b1,f_g1,f_be1,f_W2,f_b2
  const float* P[42];
  for (int i = 0; i < 42; ++i) P[i] = (const float*)d_in[4 + i];

  size_t off = 0;
  auto alloc = [&](size_t bytes) -> void* {
    void* p = (char*)d_ws + off;
    off += (bytes + 255) & ~(size_t)255;
    return p;
  };
  unsigned* aggm = (unsigned*)alloc((size_t)NN * 128 * 4);
  unsigned* gagg = (unsigned*)alloc((size_t)NG * 128 * 4);
  _Float16* feat = (_Float16*)alloc((size_t)NN * 128 * 2);
  float*    gbuf = (float*)alloc((size_t)NG * 128 * 4);
  float* b2f[4];
  for (int l = 0; l < 4; ++l) b2f[l] = (float*)alloc(128 * 4);

  auto packHalves = [](int KP, int N) { return (size_t)(N / 16) * (KP / 32) * 512; };
  _Float16* w1p[4]; _Float16* w2p[4];
  const int CINs[4] = {2, 64, 128, 128};
  const int D1s[4]  = {32, 64, 128, 128};
  const int D2s[4]  = {64, 128, 128, 128};
  for (int l = 0; l < 4; ++l) {
    int K1P = ((CINs[l] + 3) + 31) & ~31;
    w1p[l] = (_Float16*)alloc(packHalves(K1P, D1s[l]) * 2);
    w2p[l] = (_Float16*)alloc(packHalves(D1s[l], D2s[l]) * 2);
  }
  _Float16* wpp = (_Float16*)alloc(packHalves(160, 128) * 2);

  auto cdiv = [](int a, int b) { return (a + b - 1) / b; };

  // ---- prep: pack weights, fold BN1 into layer-2 weights/bias ----
  for (int l = 0; l < 4; ++l) {
    const float* W1  = P[8 * l + 0];
    const float* b2  = P[8 * l + 5];
    const float* g1  = P[8 * l + 2];
    const float* be1 = P[8 * l + 3];
    const float* W2  = P[8 * l + 4];
    int K1 = CINs[l] + 3, K1P = (K1 + 31) & ~31;
    int t1 = (int)packHalves(K1P, D1s[l]);
    int t2 = (int)packHalves(D1s[l], D2s[l]);
    pack_w<<<cdiv(t1, 256), 256, 0, stream>>>(W1, nullptr, w1p[l], K1, K1P, D1s[l], 1.0f);
    pack_w<<<cdiv(t2, 256), 256, 0, stream>>>(W2, g1, w2p[l], D1s[l], D1s[l], D2s[l], BN_RSQRT);
    fold_bias<<<1, D2s[l], 0, stream>>>(W2, b2, be1, b2f[l], D1s[l], D2s[l]);
  }
  {
    int tp = (int)packHalves(160, 128);
    pack_w<<<cdiv(tp, 256), 256, 0, stream>>>(P[32], nullptr, wpp, 131, 160, 128, 1.0f);
  }

  const int egrid = cdiv(NE, 64);

  // ---- conv1 ----
  fill_u32<<<cdiv(NN * 64, 256), 256, 0, stream>>>(aggm, NEG_INF_MAPPED, NN * 64);
  edge_mlp<2, 32, 64, 0><<<egrid, 128, 0, stream>>>(
      x, nullptr, pos, src, dst, NE, w1p[0], P[1], w2p[0], b2f[0], P[6], P[7], aggm);
  fin_feat<<<cdiv(NN * 64, 256), 256, 0, stream>>>(aggm, feat, NN * 64);

  // ---- conv2 ----
  fill_u32<<<cdiv(NN * 128, 256), 256, 0, stream>>>(aggm, NEG_INF_MAPPED, NN * 128);
  edge_mlp<64, 64, 128, 1><<<egrid, 128, 0, stream>>>(
      nullptr, feat, pos, src, dst, NE, w1p[1], P[9], w2p[1], b2f[1], P[14], P[15], aggm);
  fin_feat<<<cdiv(NN * 128, 256), 256, 0, stream>>>(aggm, feat, NN * 128);

  // ---- conv3 ----
  fill_u32<<<cdiv(NN * 128, 256), 256, 0, stream>>>(aggm, NEG_INF_MAPPED, NN * 128);
  edge_mlp<128, 128, 128, 1><<<egrid, 128, 0, stream>>>(
      nullptr, feat, pos, src, dst, NE, w1p[2], P[17], w2p[2], b2f[2], P[22], P[23], aggm);
  fin_feat<<<cdiv(NN * 128, 256), 256, 0, stream>>>(aggm, feat, NN * 128);

  // ---- conv4 ----
  fill_u32<<<cdiv(NN * 128, 256), 256, 0, stream>>>(aggm, NEG_INF_MAPPED, NN * 128);
  edge_mlp<128, 128, 128, 1><<<egrid, 128, 0, stream>>>(
      nullptr, feat, pos, src, dst, NE, w1p[3], P[25], w2p[3], b2f[3], P[30], P[31], aggm);
  fin_feat<<<cdiv(NN * 128, 256), 256, 0, stream>>>(aggm, feat, NN * 128);

  // ---- pooling net + global max pool per graph ----
  fill_u32<<<cdiv(NG * 128, 256), 256, 0, stream>>>(gagg, NEG_INF_MAPPED, NG * 128);
  edge_mlp<128, 128, 128, 2><<<cdiv(NN, 64), 128, 0, stream>>>(
      nullptr, feat, pos, nullptr, batch, NN, wpp, P[33], nullptr, nullptr, P[34], P[35], gagg);
  fin_graph<<<cdiv(NG * 128, 256), 256, 0, stream>>>(gagg, gbuf, NG * 128);

  // ---- FC head ----
  head_kernel<<<1, 256, 0, stream>>>(gbuf, P[36], P[37], P[38], P[39], P[40], P[41],
                                     (float*)d_out);
}
